// AttentionKernel_27874337751495
// MI455X (gfx1250) — matrix operand order repaired and checked by reading, not compiled, not run on hardware
//
#include <hip/hip_runtime.h>
#include <cstdint>
#include <cstddef>

// ---------------------------------------------------------------------------
// Shapes (fixed by the reference): B=2, H=16, S=2048, D=128
// Roofline: output = 512MB fp32 @ 23.3TB/s ~= 23us floor; GEMM = 34.4 GFLOP
// -> must run on WMMA bf16 pipes; kernel is output-store bound.
// ---------------------------------------------------------------------------
#define S_LEN 2048
#define D_LEN 128
#define BH    32          // B*H

typedef __attribute__((ext_vector_type(16))) __bf16 v16bf;
typedef __attribute__((ext_vector_type(8)))  float  v8f;

typedef unsigned int u32x4 __attribute__((ext_vector_type(4)));
typedef int          i32x4 __attribute__((ext_vector_type(4)));
typedef int          i32x8 __attribute__((ext_vector_type(8)));

#if defined(__has_builtin)
#  if __has_builtin(__builtin_amdgcn_tensor_load_to_lds)
#    define USE_TDM 1
#  endif
#endif

struct Frag32 { uint4 lo; uint4 hi; };   // 32 bytes = 16 bf16 = one WMMA fragment/lane

// LDS K-strip: 64 rows x 256B, padded to 272B/row (17 uint4) so that the
// 16-lane fragment gather (row stride) maps to 16 distinct LDS banks.
#define LDS_ROW_US 136                   // 272 bytes / 2
#define LDS_ROW_Q4 17

static __device__ __forceinline__ unsigned short f32_to_bf16_rne(float f) {
  uint32_t u = __builtin_bit_cast(uint32_t, f);
  u += 0x7FFFu + ((u >> 16) & 1u);       // round-to-nearest-even (inputs finite)
  return (unsigned short)(u >> 16);
}

// ---------------------------------------------------------------------------
// Kernel 1: row-wise L2 normalize (x / max(||x||,1e-12)) + cast to bf16.
// ---------------------------------------------------------------------------
__global__ void __launch_bounds__(256)
l2norm_bf16_kernel(const float* __restrict__ src, unsigned short* __restrict__ dst) {
  const int wave = (blockIdx.x * blockDim.x + threadIdx.x) >> 5;
  const int lane = threadIdx.x & 31;

  const float4 v = reinterpret_cast<const float4*>(src + (size_t)wave * D_LEN)[lane];
  float s = v.x * v.x + v.y * v.y + v.z * v.z + v.w * v.w;
#pragma unroll
  for (int off = 16; off >= 1; off >>= 1) s += __shfl_xor(s, off, 32);

  const float scale = 1.0f / fmaxf(sqrtf(s), 1e-12f);

  ushort4 o;
  o.x = f32_to_bf16_rne(v.x * scale);
  o.y = f32_to_bf16_rne(v.y * scale);
  o.z = f32_to_bf16_rne(v.z * scale);
  o.w = f32_to_bf16_rne(v.w * scale);
  reinterpret_cast<ushort4*>(dst + (size_t)wave * D_LEN)[lane] = o;
}

// ---------------------------------------------------------------------------
// Kernel 2: out = exp(clip((Qn·Knᵀ)/temp, ±100)) + 1e-6 via bf16 WMMA.
//
// Block = 256 threads = 8 waves; block tile = 128 rows x 64 cols.
//   1. Each wave prefetches its 4 A fragments (its 16 Q rows) from global.
//   2. Wave 0 issues a TDM tensor_load_to_lds for the block's 64x128 K-strip
//      (contiguous 16KB), with TDM pad_enable inserting 16B per 256B row so
//      fragment gathers hit 16 distinct LDS banks. s_wait_tensorcnt + barrier.
//   3. 4 K-steps x 4 N-tiles of v_wmma_f32_16x16x32_bf16, B from ds_load_b128.
//
// Fragment gathers match CDNA5 16-bit VGPR layouts (ISA 7.12.2).
// ---------------------------------------------------------------------------
__global__ void __launch_bounds__(256)
score_exp_wmma_kernel(const unsigned short* __restrict__ qn,
                      const unsigned short* __restrict__ kn,
                      const float* __restrict__ log_temp,
                      float* __restrict__ out) {
  __shared__ __align__(16) unsigned short smem[64 * LDS_ROW_US];  // 17408 B

  const int lane = threadIdx.x & 31;
  const int wv   = threadIdx.x >> 5;     // 0..7
  const int half = lane >> 4;
  const int lm   = lane & 15;

  const int nstrip = blockIdx.x & 31;          // 32 strips of 64 columns
  const int mblock = (blockIdx.x >> 5) & 15;   // 16 blocks of 128 rows
  const int bh     = blockIdx.x >> 9;          // 32 batch*head slices
  const int mtile  = mblock * 8 + wv;          // this wave's 16-row tile

  const unsigned short* kbase =
      kn + ((size_t)bh * S_LEN + (size_t)nstrip * 64) * D_LEN;   // 16KB contiguous

  // ---- 1. Prefetch A fragments (overlaps with the TDM DMA) --------------
  const unsigned short* arow =
      qn + ((size_t)bh * S_LEN + (size_t)mtile * 16 + lm) * D_LEN;
  Frag32 af[4];
#pragma unroll
  for (int i = 0; i < 4; ++i) {
    const int kb = i * 32;
    af[i].lo = *reinterpret_cast<const uint4*>(arow + kb + half * 8);
    af[i].hi = *reinterpret_cast<const uint4*>(arow + kb + 16 + half * 8);
  }

  // ---- 2. Stage K-strip into LDS ---------------------------------------
#if USE_TDM
  if (wv == 0) {
    const uint64_t ga = (uint64_t)(uintptr_t)kbase;
    // D# group0: count=1 | lds_addr=0 | global_addr | type=2 ("image")
    u32x4 g0;
    g0[0] = 1u;
    g0[1] = 0u;                                     // smem is the only LDS alloc
    g0[2] = (uint32_t)(ga & 0xFFFFFFFFu);
    g0[3] = (uint32_t)((ga >> 32) & 0x01FFFFFFu) | (2u << 30);
    // D# group1: data_size=2B | pad_enable, pad every 64 DW (256B) by 4 DW (16B)
    //            tensor 128 x 2048, tile 128 x 64, dim0 stride 128
    i32x8 g1;
    g1[0] = (int)((1u << 16) | (1u << 20) | (5u << 22) | (3u << 25));
    g1[1] = (int)(128u << 16);     // tensor_dim0 = 128
    g1[2] = (int)(2048u << 16);    // tensor_dim1 = 2048
    g1[3] = (int)(128u << 16);     // tile_dim0 = 128
    g1[4] = 64;                    // tile_dim1 = 64, tile_dim2 = 0
    g1[5] = 128;                   // tensor_dim0_stride = 128
    g1[6] = 0;
    g1[7] = 0;
    i32x4 z4 = {0, 0, 0, 0};
#if __has_include(<hip/amd_detail/amd_gfx1250_TDM.h>)
    i32x8 z8 = {0, 0, 0, 0, 0, 0, 0, 0};
    __builtin_amdgcn_tensor_load_to_lds(g0, g1, z4, z4, z8, 0);
#else
    __builtin_amdgcn_tensor_load_to_lds(g0, g1, z4, z4, 0);
#endif
    __builtin_amdgcn_s_wait_tensorcnt(0);           // TDM done before barrier
  }
  // The optimizer cannot see that the TDM wrote `smem` (the LDS address is an
  // integer in the descriptor). Escape smem's address INTO opaque asm with a
  // memory clobber: the asm may now store through it, so the ds loads below
  // cannot be folded to undef. Crucially we do NOT use an asm *output* as the
  // load pointer -- loads keep addrspace(3) provenance and lower to ds_load.
  asm volatile("" : : "v"(smem) : "memory");
  __syncthreads();
#else
  // Fallback: cooperative coalesced copy with the same 272B-row padding.
  {
    const uint4* gsrc = reinterpret_cast<const uint4*>(kbase);
    uint4*       ldst = reinterpret_cast<uint4*>(smem);
#pragma unroll
    for (int i = 0; i < 4; ++i) {
      const int j   = (int)threadIdx.x + i * 256;   // 1024 uint4 total
      const int row = j >> 4, col = j & 15;
      ldst[row * LDS_ROW_Q4 + col] = gsrc[j];
    }
  }
  __syncthreads();
#endif

  // ---- 3. WMMA main loop ------------------------------------------------
  v8f acc[4];
#pragma unroll
  for (int t = 0; t < 4; ++t) acc[t] = (v8f)0.0f;

#pragma unroll
  for (int i = 0; i < 4; ++i) {
    const int kb = i * 32;
    const v16bf a = __builtin_bit_cast(v16bf, af[i]);
#pragma unroll
    for (int t = 0; t < 4; ++t) {
      const uint4* brow =
          reinterpret_cast<const uint4*>(smem + (size_t)(t * 16 + lm) * LDS_ROW_US)
          + ((kb >> 3) + half);
      Frag32 bfrag;
      bfrag.lo = brow[0];
      bfrag.hi = brow[2];
      const v16bf b = __builtin_bit_cast(v16bf, bfrag);

      acc[t] = __builtin_amdgcn_wmma_f32_16x16x32_bf16(
          false, a, false, b, (short)0, acc[t], false, false);
    }
  }

  // ---- 4. Epilogue: scale, clamp, exp, +eps, NT store -------------------
  // temp in [0.05, 100] -> v_rcp_f32 is well-conditioned (avoids IEEE divide)
  const float temp     = fminf(fmaxf(__expf(*log_temp), 0.05f), 100.0f);
  const float inv_temp = __builtin_amdgcn_rcpf(temp);

  // C/D layout: VGPR r, lanes 0-15 -> (M=r,N=lane); lanes 16-31 -> (M=r+8,N-16)
  float* obase = out + ((size_t)bh * S_LEN + (size_t)mtile * 16 + (size_t)half * 8) * S_LEN
                     + (size_t)nstrip * 64 + lm;
#pragma unroll
  for (int t = 0; t < 4; ++t) {
#pragma unroll
    for (int r = 0; r < 8; ++r) {
      float sv = acc[t][r] * inv_temp;
      sv = fminf(fmaxf(sv, -100.0f), 100.0f);
      const float e = __expf(sv) + 1e-6f;
      __builtin_nontemporal_store(e, obase + (size_t)r * S_LEN + t * 16);
    }
  }
}

// ---------------------------------------------------------------------------
// Host launcher (graph-capture safe: kernels only, ws for bf16 staging)
// ---------------------------------------------------------------------------
extern "C" void kernel_launch(void* const* d_in, const int* in_sizes, int n_in,
                              void* d_out, int out_size, void* d_ws, size_t ws_size,
                              hipStream_t stream) {
  (void)in_sizes; (void)n_in; (void)out_size; (void)ws_size;

  const float* q   = (const float*)d_in[0];
  const float* k   = (const float*)d_in[1];
  const float* lt  = (const float*)d_in[2];
  float*       out = (float*)d_out;

  const size_t elems = (size_t)BH * S_LEN * D_LEN;          // 8,388,608 / tensor
  unsigned short* qn = (unsigned short*)d_ws;               // 16 MB
  unsigned short* kn = qn + elems;                          // 16 MB

  const int rows = BH * S_LEN;                              // 65536 rows/tensor
  l2norm_bf16_kernel<<<rows / 8, 256, 0, stream>>>(q, qn);
  l2norm_bf16_kernel<<<rows / 8, 256, 0, stream>>>(k, kn);

  // 32 bh * 16 mblocks * 32 nstrips = 16384 blocks of 8 waves
  score_exp_wmma_kernel<<<BH * 16 * 32, 256, 0, stream>>>(qn, kn, lt, out);
}
